// Simple3d_72962904425001
// MI455X (gfx1250) — matrix operand order repaired and checked by reading, not compiled, not run on hardware
//
#include <hip/hip_runtime.h>

// ---------------------------------------------------------------------------
// Sparse submanifold conv stack for MI455X (gfx1250, wave32).
// Strategy: f16 activations (AoS [voxel][16ch]) + v_wmma_f32_16x16x32_f16.
// Each wave: D[16 outch x 16 voxels] accumulated over 27 taps packed 2-per-WMMA.
// BN stats (sum/sumsq over active voxels) accumulated single-pass via
// shuffle + LDS + global float atomics; folded into scale/shift applied by the
// consumer kernel while staging its LDS input tile.
// ---------------------------------------------------------------------------

#define DHW   128
#define VOX   (128 * 128 * 128)
#define SLOPE 0.01f
#define EPS   1e-5f

// conv tile
#define TW 32
#define THH 4
#define TDD 4
#define IW 34
#define IH 6
#define ID 6
#define NSLOT (ID * IH * IW)   // 1224 input-tile voxels (incl. halo)

typedef _Float16 v16h __attribute__((ext_vector_type(16)));
typedef _Float16 v8h  __attribute__((ext_vector_type(8)));
typedef float    v8f  __attribute__((ext_vector_type(8)));

__device__ __forceinline__ float leaky(float x) { return x >= 0.f ? x : SLOPE * x; }

// ---------------------------------------------------------------------------
__global__ __launch_bounds__(256) void k_zero(float* p, int n) {
  int i = blockIdx.x * 256 + threadIdx.x;
  if (i < n) p[i] = 0.f;
}

// ---------------------------------------------------------------------------
// Layer 0: 1x1 conv 4->16 (+bias, mask), stats, active count, maskf output.
__global__ __launch_bounds__(256) void k_layer0(
    const float* __restrict__ feat, const int* __restrict__ mask,
    const float* __restrict__ w0, const float* __restrict__ b0,
    _Float16* __restrict__ y0, float* __restrict__ cnt,
    float* __restrict__ st /*[0..15]=sum,[16..31]=sumsq*/,
    float* __restrict__ maskOut) {
  __shared__ float s_sum[16], s_sq[16], s_cnt;
  int tid = threadIdx.x;
  if (tid < 16) { s_sum[tid] = 0.f; s_sq[tid] = 0.f; }
  if (tid == 0) s_cnt = 0.f;
  __syncthreads();

  int v = blockIdx.x * 256 + tid;
  float mf = (mask[v] > 0) ? 1.f : 0.f;
  float x0 = feat[v], x1 = feat[VOX + v], x2 = feat[2 * VOX + v], x3 = feat[3 * VOX + v];

  float ys[16];
#pragma unroll
  for (int o = 0; o < 16; ++o) {
    float a = w0[o * 4 + 0] * x0 + w0[o * 4 + 1] * x1 +
              w0[o * 4 + 2] * x2 + w0[o * 4 + 3] * x3 + b0[o];
    ys[o] = a * mf;
  }

  // store raw (pre-BN) activations as f16 AoS [voxel][16]
  v8h lo, hi;
#pragma unroll
  for (int c = 0; c < 8; ++c) { lo[c] = (_Float16)ys[c]; hi[c] = (_Float16)ys[8 + c]; }
  *(v8h*)(y0 + v * 16)     = lo;
  *(v8h*)(y0 + v * 16 + 8) = hi;

  maskOut[v] = mf;

  // wave-level reduce, then LDS, then global atomics
#pragma unroll
  for (int c = 0; c < 16; ++c) {
    float a = ys[c], q = ys[c] * ys[c];
    for (int off = 16; off >= 1; off >>= 1) {
      a += __shfl_xor(a, off, 32);
      q += __shfl_xor(q, off, 32);
    }
    if ((tid & 31) == 0) { atomicAdd(&s_sum[c], a); atomicAdd(&s_sq[c], q); }
  }
  float cl = mf;
  for (int off = 16; off >= 1; off >>= 1) cl += __shfl_xor(cl, off, 32);
  if ((tid & 31) == 0) atomicAdd(&s_cnt, cl);

  __syncthreads();
  if (tid < 16) {
    atomicAdd(&st[tid], s_sum[tid]);
    atomicAdd(&st[16 + tid], s_sq[tid]);
  }
  if (tid == 16) atomicAdd(cnt, s_cnt);
}

// ---------------------------------------------------------------------------
// Fold BN stats into per-channel scale/shift.
__global__ void k_bnfin(const float* __restrict__ cnt, const float* __restrict__ st,
                        const float* __restrict__ gam, const float* __restrict__ bet,
                        float* __restrict__ sc, float* __restrict__ sh) {
  int c = threadIdx.x;
  if (c >= 16) return;
  float n = fmaxf(cnt[0], 1.f);
  float mean = st[c] / n;
  float var = st[16 + c] / n - mean * mean;
  float s = gam[c] * rsqrtf(var + EPS);
  sc[c] = s;
  sh[c] = bet[c] - mean * s;
}

// ---------------------------------------------------------------------------
// 3x3x3 subm conv 16->16 via WMMA. Input = bn+leaky(prev raw) applied in the
// LDS staging pass; output = raw (conv+bias)*mask, stats accumulated.
__global__ __launch_bounds__(256) void k_conv3(
    const _Float16* __restrict__ yin, const int* __restrict__ mask,
    const float* __restrict__ w, const float* __restrict__ bias,
    const float* __restrict__ sc, const float* __restrict__ sh,
    _Float16* __restrict__ yout, float* __restrict__ st) {
  __shared__ __align__(16) _Float16 zl[(NSLOT + 1) * 16];  // +1 zero slot (tap pad)
  __shared__ __align__(16) _Float16 wl[28 * 16 * 16];      // tap 27 zeroed
  __shared__ float s_bn[48];   // [0..15]=scale [16..31]=shift [32..47]=bias
  __shared__ float s_st[32];

  int tid = threadIdx.x;
  if (tid < 16) { s_bn[tid] = sc[tid]; s_bn[16 + tid] = sh[tid]; s_bn[32 + tid] = bias[tid]; }
  if (tid < 32) s_st[tid] = 0.f;

  // weights -> LDS as wl[(tap*16+o)*16+i], f16; tap 27 zero pad
  for (int idx = tid; idx < 6912; idx += 256) {
    int tap = idx >> 8, rem = idx & 255, o = rem >> 4, i = rem & 15;
    wl[idx] = (_Float16)w[o * 432 + i * 27 + tap];
  }
  wl[6912 + tid] = (_Float16)0;  // 256 elems, one per thread
  if (tid < 16) zl[NSLOT * 16 + tid] = (_Float16)0;
  __syncthreads();

  int bx = blockIdx.x;
  int twi = bx & 3, thi = (bx >> 2) & 31, tdi = bx >> 7;
  int w0p = twi * TW, h0 = thi * THH, d0 = tdi * TDD;

  // stage input tile (with halo) into LDS, applying BN + LeakyReLU + mask
  for (int s = tid; s < NSLOT; s += 256) {
    int dd = s / (IH * IW);
    int r2 = s - dd * (IH * IW);
    int hh = r2 / IW, ww = r2 - hh * IW;
    int gd = d0 - 1 + dd, gh = h0 - 1 + hh, gw = w0p - 1 + ww;
    v8h lo = {}, hi = {};
    if ((unsigned)gd < 128u && (unsigned)gh < 128u && (unsigned)gw < 128u) {
      int vox = (gd * 128 + gh) * 128 + gw;
      if (mask[vox] > 0) {
        v8h a = *(const v8h*)(yin + vox * 16);
        v8h b = *(const v8h*)(yin + vox * 16 + 8);
#pragma unroll
        for (int c = 0; c < 8; ++c) {
          lo[c] = (_Float16)leaky(s_bn[c] * (float)a[c] + s_bn[16 + c]);
          hi[c] = (_Float16)leaky(s_bn[8 + c] * (float)b[c] + s_bn[24 + c]);
        }
      }
    }
    *(v8h*)(zl + s * 16)     = lo;
    *(v8h*)(zl + s * 16 + 8) = hi;
  }
  __syncthreads();

  int lane = tid & 31, wv = tid >> 5;
  int n = lane & 15, g = lane >> 4;   // n = voxel col = A row; g = half select
  float sacc[8] = {0, 0, 0, 0, 0, 0, 0, 0};
  float sqacc[8] = {0, 0, 0, 0, 0, 0, 0, 0};

  for (int nt = wv; nt < 32; nt += 8) {
    int dd = nt >> 3, hh = (nt >> 1) & 3, wseg = nt & 1;
    int wbase = wseg * 16;
    v8f acc = {};
#pragma unroll
    for (int p = 0; p < 14; ++p) {
      const int t0 = 2 * p, t1 = 2 * p + 1;
      // A fragment: lane row n; half g holds w[g ? t1 : t0][n][0..15]
      v8h a0 = *(const v8h*)(wl + ((g ? t1 : t0) * 16 + n) * 16);
      v8h a1 = *(const v8h*)(wl + ((g ? t1 : t0) * 16 + n) * 16 + 8);
      v16h A;
#pragma unroll
      for (int i = 0; i < 8; ++i) { A[i] = a0[i]; A[8 + i] = a1[i]; }
      // B fragment: lane reads all 16 channels at one (tap, voxel) position
      const int kd0 = t0 / 9, kh0 = (t0 / 3) % 3, kw0 = t0 % 3;
      int slot0 = ((dd + kd0) * IH + (hh + kh0)) * IW + wbase + n + kw0;
      int slot1;
      if (t1 < 27) {
        const int kd1 = t1 / 9, kh1 = (t1 / 3) % 3, kw1 = t1 % 3;
        slot1 = ((dd + kd1) * IH + (hh + kh1)) * IW + wbase + n + kw1;
      } else {
        slot1 = NSLOT;  // zero slot
      }
      int slot = g ? slot1 : slot0;
      v8h b0v = *(const v8h*)(zl + slot * 16);
      v8h b1v = *(const v8h*)(zl + slot * 16 + 8);
      v16h B;
#pragma unroll
      for (int i = 0; i < 8; ++i) { B[i] = b0v[i]; B[8 + i] = b1v[i]; }

      acc = __builtin_amdgcn_wmma_f32_16x16x32_f16(false, A, false, B,
                                                   (short)0, acc, false, false);
    }

    // epilogue: +bias, mask, store raw f16, accumulate stats
    int gw_ = w0p + wbase + n;
    int vox = ((d0 + dd) * 128 + (h0 + hh)) * 128 + gw_;
    float mf = (mask[vox] > 0) ? 1.f : 0.f;
    v8h outv;
#pragma unroll
    for (int r = 0; r < 8; ++r) {
      int c = r + 8 * g;
      float val = (acc[r] + s_bn[32 + c]) * mf;
      outv[r] = (_Float16)val;
      sacc[r] += val;
      sqacc[r] += val * val;
    }
    *(v8h*)(yout + vox * 16 + 8 * g) = outv;
  }

  // reduce over the 16 lanes sharing a channel group (xor 1,2,4,8 stays in group)
#pragma unroll
  for (int r = 0; r < 8; ++r) {
    float a = sacc[r], q = sqacc[r];
    for (int off = 8; off >= 1; off >>= 1) {
      a += __shfl_xor(a, off, 32);
      q += __shfl_xor(q, off, 32);
    }
    if (n == 0) {
      atomicAdd(&s_st[r + 8 * g], a);
      atomicAdd(&s_st[16 + r + 8 * g], q);
    }
  }
  __syncthreads();
  if (tid < 32) atomicAdd(&st[tid], s_st[tid]);
}

// ---------------------------------------------------------------------------
// Final: BN4+LeakyReLU on y4, 1x1 conv 16->16, mask, NCDHW f32 output.
__global__ __launch_bounds__(256) void k_final(
    const _Float16* __restrict__ y4, const int* __restrict__ mask,
    const float* __restrict__ w5, const float* __restrict__ b5,
    const float* __restrict__ sc, const float* __restrict__ sh,
    float* __restrict__ out) {
  __shared__ float s_w[256], s_b[16], s_sc[16], s_sh[16];
  int tid = threadIdx.x;
  s_w[tid] = w5[tid];
  if (tid < 16) { s_b[tid] = b5[tid]; s_sc[tid] = sc[tid]; s_sh[tid] = sh[tid]; }
  __syncthreads();

  int v = blockIdx.x * 256 + tid;
  float mf = (mask[v] > 0) ? 1.f : 0.f;
  float z[16];
  v8h a = *(const v8h*)(y4 + v * 16);
  v8h b = *(const v8h*)(y4 + v * 16 + 8);
#pragma unroll
  for (int c = 0; c < 8; ++c) {
    z[c]     = mf != 0.f ? leaky(s_sc[c] * (float)a[c] + s_sh[c]) : 0.f;
    z[8 + c] = mf != 0.f ? leaky(s_sc[8 + c] * (float)b[c] + s_sh[8 + c]) : 0.f;
  }
#pragma unroll
  for (int o = 0; o < 16; ++o) {
    float acc = s_b[o];
#pragma unroll
    for (int c = 0; c < 16; ++c) acc += s_w[o * 16 + c] * z[c];
    out[o * VOX + v] = acc * mf;
  }
}

// ---------------------------------------------------------------------------
extern "C" void kernel_launch(void* const* d_in, const int* in_sizes, int n_in,
                              void* d_out, int out_size, void* d_ws, size_t ws_size,
                              hipStream_t stream) {
  (void)in_sizes; (void)n_in; (void)out_size; (void)ws_size;

  const float* feat = (const float*)d_in[0];
  const int*   mask = (const int*)d_in[1];
  const float* w0 = (const float*)d_in[2];
  const float* b0 = (const float*)d_in[3];
  const float* g0 = (const float*)d_in[4];
  const float* be0 = (const float*)d_in[5];
  const float* wc[4], *bc[4], *gc[4], *bec[4];
  for (int i = 0; i < 4; ++i) {
    wc[i]  = (const float*)d_in[6 + 4 * i];
    bc[i]  = (const float*)d_in[7 + 4 * i];
    gc[i]  = (const float*)d_in[8 + 4 * i];
    bec[i] = (const float*)d_in[9 + 4 * i];
  }
  const float* w5 = (const float*)d_in[22];
  const float* b5 = (const float*)d_in[23];

  char* ws = (char*)d_ws;
  const size_t SZY = (size_t)VOX * 16 * sizeof(_Float16);  // 64 MB
  _Float16* yA = (_Float16*)ws;
  _Float16* yB = (_Float16*)(ws + SZY);
  float* stats = (float*)(ws + 2 * SZY);
  // stats[0]=active count; per-layer L: sums at 1+32L (16 sum + 16 sumsq),
  // folded params at 161+32L (16 scale + 16 shift)
  float* cnt = stats;
  float* out = (float*)d_out;

  auto lsum = [&](int L) { return stats + 1 + 32 * L; };
  auto lpar = [&](int L) { return stats + 161 + 32 * L; };

  k_zero<<<2, 256, 0, stream>>>(stats, 321);

  k_layer0<<<VOX / 256, 256, 0, stream>>>(feat, mask, w0, b0, yA, cnt, lsum(0),
                                          out + (size_t)16 * VOX);
  k_bnfin<<<1, 16, 0, stream>>>(cnt, lsum(0), g0, be0, lpar(0), lpar(0) + 16);

  _Float16* src = yA;
  _Float16* dst = yB;
  for (int L = 0; L < 4; ++L) {
    k_conv3<<<4096, 256, 0, stream>>>(src, mask, wc[L], bc[L],
                                      lpar(L), lpar(L) + 16, dst, lsum(L + 1));
    k_bnfin<<<1, 16, 0, stream>>>(cnt, lsum(L + 1), gc[L], bec[L],
                                  lpar(L + 1), lpar(L + 1) + 16);
    _Float16* t = src; src = dst; dst = t;
  }

  // after 4 swaps, src holds y4 raw
  k_final<<<VOX / 256, 256, 0, stream>>>(src, mask, w5, b5,
                                         lpar(4), lpar(4) + 16, out);
}
